// EncoderLayer_80917183856671
// MI455X (gfx1250) — matrix operand order repaired and checked by reading, not compiled, not run on hardware
//
#include <hip/hip_runtime.h>
#include <hip/hip_bf16.h>
#include <math.h>

#define B_   2
#define S_   2048
#define D_   1024
#define H_   16
#define DK_  64
#define DF_  4096
#define NTOK (B_*S_)

typedef __attribute__((ext_vector_type(16))) __bf16 v16bf;
typedef __attribute__((ext_vector_type(8)))  __bf16 v8bf;
typedef __attribute__((ext_vector_type(8)))  float  v8f;

static __device__ __forceinline__ v16bf comb16(v8bf lo, v8bf hi) {
  return __builtin_shufflevector(lo, hi, 0,1,2,3,4,5,6,7,8,9,10,11,12,13,14,15);
}
static __device__ __forceinline__ v8f wmma_bf16(v16bf a, v16bf b, v8f c) {
  return __builtin_amdgcn_wmma_f32_16x16x32_bf16(false, a, false, b, (short)0, c, false, false);
}
static __device__ __forceinline__ float redmax16(float v) {
  v = fmaxf(v, __shfl_xor(v, 1, 32));
  v = fmaxf(v, __shfl_xor(v, 2, 32));
  v = fmaxf(v, __shfl_xor(v, 4, 32));
  v = fmaxf(v, __shfl_xor(v, 8, 32));
  return v;
}
static __device__ __forceinline__ float redsum16(float v) {
  v += __shfl_xor(v, 1, 32);
  v += __shfl_xor(v, 2, 32);
  v += __shfl_xor(v, 4, 32);
  v += __shfl_xor(v, 8, 32);
  return v;
}

// CDNA5 async global->LDS copy, 16 bytes per lane, tracked by ASYNCcnt.
// lds = LDS byte address (low 32 bits of flat shared pointer), g = global address.
static __device__ __forceinline__ void async_cp16(unsigned lds, const __bf16* g) {
  asm volatile("global_load_async_to_lds_b128 %0, %1, off"
               :: "v"(lds), "v"((unsigned long long)g) : "memory");
}
static __device__ __forceinline__ void wait_async0() {
  asm volatile("s_wait_asynccnt 0x0" ::: "memory");
}
static __device__ __forceinline__ unsigned lds_off(const void* p) {
  return (unsigned)(unsigned long long)p;
}

// ---------------- LayerNorm: fp32 in -> bf16 out ----------------
__global__ __launch_bounds__(256)
void ln_kernel(const float* __restrict__ x, const float* __restrict__ g,
               const float* __restrict__ be, __bf16* __restrict__ out) {
  __shared__ float red[256];
  int tok = blockIdx.x;
  int t = threadIdx.x;
  const float* xr = x + (size_t)tok * D_;
  float4 v = *(const float4*)(xr + t * 4);
  red[t] = v.x + v.y + v.z + v.w;
  __syncthreads();
  for (int o = 128; o > 0; o >>= 1) { if (t < o) red[t] += red[t + o]; __syncthreads(); }
  float mu = red[0] * (1.0f / D_);
  __syncthreads();
  float dx = v.x - mu, dy = v.y - mu, dz = v.z - mu, dw = v.w - mu;
  red[t] = dx*dx + dy*dy + dz*dz + dw*dw;
  __syncthreads();
  for (int o = 128; o > 0; o >>= 1) { if (t < o) red[t] += red[t + o]; __syncthreads(); }
  float rstd = rsqrtf(red[0] * (1.0f / D_) + 1e-5f);
  int c = t * 4;
  __bf16* orow = out + (size_t)tok * D_;
  orow[c + 0] = (__bf16)(dx * rstd * g[c + 0] + be[c + 0]);
  orow[c + 1] = (__bf16)(dy * rstd * g[c + 1] + be[c + 1]);
  orow[c + 2] = (__bf16)(dz * rstd * g[c + 2] + be[c + 2]);
  orow[c + 3] = (__bf16)(dw * rstd * g[c + 3] + be[c + 3]);
}

// ---------------- Weight converts ----------------
__global__ __launch_bounds__(256)
void conv_wqkv(const float* __restrict__ Wq, const float* __restrict__ Wk,
               const float* __restrict__ Wv, __bf16* __restrict__ wT) {
  size_t idx = (size_t)blockIdx.x * 256 + threadIdx.x;   // 3072*1024 total
  int nrow = (int)(idx >> 10), d = (int)(idx & 1023);
  int mat = nrow >> 10, n2 = nrow & 1023;
  int hh = n2 >> 6, kk = n2 & 63;
  const float* W = (mat == 0) ? Wq : (mat == 1) ? Wk : Wv;
  float v = W[((size_t)hh * D_ + d) * DK_ + kk];
  if (mat == 0) v *= 0.125f;
  wT[idx] = (__bf16)v;
}

__global__ __launch_bounds__(256)
void conv_bqkv(const float* __restrict__ bq, const float* __restrict__ bk,
               const float* __restrict__ bv, float* __restrict__ bqkv) {
  int i = blockIdx.x * 256 + threadIdx.x;   // 3072
  float v = (i < 1024) ? bq[i] * 0.125f : (i < 2048) ? bk[i - 1024] : bv[i - 2048];
  bqkv[i] = v;
}

// out[n*K + k] = bf16(in[k*N + n])
__global__ __launch_bounds__(256)
void conv_T(const float* __restrict__ in, __bf16* __restrict__ out, int N, int K) {
  size_t idx = (size_t)blockIdx.x * 256 + threadIdx.x;   // N*K total
  int n = (int)(idx / K), k2 = (int)(idx % K);
  out[idx] = (__bf16)(in[(size_t)k2 * N + n]);
}

// ---------------- WMMA bf16 GEMM: C = A[M,K] x Bt[N,K]^T ----------------
// Block tile 64(M) x 128(N), 4 waves, wave tile 32x64 (2x4 WMMA frags).
// Double-buffered LDS tiles filled with global_load_async_to_lds_b128.
// epi: 0=QKV scatter(+bias), 1/3=bias+resid->fp32, 2=FFN1(+bias,gelu->bf16)
__global__ __launch_bounds__(128)
void gemm_bf16(const __bf16* __restrict__ A, const __bf16* __restrict__ Bt,
               int M, int N, int K, int epi,
               const float* __restrict__ bias, const float* __restrict__ resid,
               float* __restrict__ outf, __bf16* __restrict__ outb,
               __bf16* __restrict__ qout, __bf16* __restrict__ kout,
               __bf16* __restrict__ vout) {
  __shared__ __bf16 As[2][64 * 40];
  __shared__ __bf16 Bs[2][128 * 40];
  int m0 = blockIdx.y * 64, n0 = blockIdx.x * 128;
  int t = threadIdx.x, lane = t & 31, wave = t >> 5;
  int wm = wave >> 1, wn = wave & 1;
  int lm = lane & 15, g16 = lane >> 4;
  int hk = g16 * 8, kb = g16 * 16;
  v8f acc[2][4] = {};

  // per-thread staging addresses: A row t>>1 (half-row chunks), B row t (full row)
  int ar = t >> 1, ac = (t & 1) * 16;
  const __bf16* gA = A + (size_t)(m0 + ar) * K + ac;
  const __bf16* gB = Bt + (size_t)(n0 + t) * K;
  unsigned lA0 = lds_off(&As[0][ar * 40 + ac]);
  unsigned lA1 = lds_off(&As[1][ar * 40 + ac]);
  unsigned lB0 = lds_off(&Bs[0][t * 40]);
  unsigned lB1 = lds_off(&Bs[1][t * 40]);

#define STAGE(lA, lB, kt)                                                     \
  do {                                                                        \
    async_cp16((lA), gA + (kt));                                              \
    async_cp16((lA) + 16, gA + (kt) + 8);                                     \
    async_cp16((lB), gB + (kt));                                              \
    async_cp16((lB) + 16, gB + (kt) + 8);                                     \
    async_cp16((lB) + 32, gB + (kt) + 16);                                    \
    async_cp16((lB) + 48, gB + (kt) + 24);                                    \
  } while (0)

  STAGE(lA0, lB0, 0);
  int buf = 0;
  for (int kt = 0; kt < K; kt += 32) {
    wait_async0();        // my async copies for current buf complete
    __syncthreads();      // everyone's complete; prev compute done (program order)
    if (kt + 32 < K) {    // prefetch next tile into other buffer
      if (buf == 0) STAGE(lA1, lB1, kt + 32);
      else          STAGE(lA0, lB0, kt + 32);
    }
    v16bf af[2], bfr[4];
#pragma unroll
    for (int i = 0; i < 2; i++) {
      const __bf16* p = &As[buf][(wm * 32 + i * 16 + lm) * 40];
      af[i] = comb16(*(const v8bf*)(p + 2 * hk), *(const v8bf*)(p + 2 * hk + 8));
    }
#pragma unroll
    for (int j = 0; j < 4; j++) {
      const __bf16* p = &Bs[buf][(wn * 64 + j * 16 + lm) * 40 + kb];
      bfr[j] = comb16(*(const v8bf*)p, *(const v8bf*)(p + 8));
    }
#pragma unroll
    for (int i = 0; i < 2; i++)
#pragma unroll
      for (int j = 0; j < 4; j++)
        acc[i][j] = wmma_bf16(af[i], bfr[j], acc[i][j]);
    buf ^= 1;
  }
#undef STAGE

#pragma unroll
  for (int i = 0; i < 2; i++) {
    int mrow0 = m0 + wm * 32 + i * 16 + g16 * 8;
#pragma unroll
    for (int j = 0; j < 4; j++) {
      int col = n0 + wn * 64 + j * 16 + lm;
#pragma unroll
      for (int r = 0; r < 8; r++) {
        int row = mrow0 + r;
        float val = acc[i][j][r];
        if (epi == 0) {
          val += bias[col];
          int mat = col >> 10, n2 = col & 1023;
          int hh = n2 >> 6, dd = n2 & 63;
          int bb = row >> 11, ss = row & 2047;
          size_t dst = ((size_t)(bb * H_ + hh) * S_ + ss) * DK_ + dd;
          __bf16 bvv = (__bf16)val;
          if (mat == 0) qout[dst] = bvv;
          else if (mat == 1) kout[dst] = bvv;
          else vout[dst] = bvv;
        } else if (epi == 2) {
          val += bias[col];
          val = 0.5f * val * (1.0f + erff(val * 0.70710678118654752f));
          outb[(size_t)row * N + col] = (__bf16)val;
        } else {  // 1 or 3: bias + residual -> fp32
          val += bias[col] + resid[(size_t)row * N + col];
          outf[(size_t)row * N + col] = val;
        }
      }
    }
  }
}

// ---------------- Flash attention ----------------
// grid (S/64, H, B), block 128 = 4 waves; wave w owns 16 query rows.
__global__ __launch_bounds__(128)
void fa_kernel(const __bf16* __restrict__ q, const __bf16* __restrict__ k,
               const __bf16* __restrict__ v, __bf16* __restrict__ ob) {
  __shared__ __bf16 VTs[64 * 72];      // [dk][key]
  __shared__ __bf16 Ps[4][16 * 72];    // per wave [row][key]
  int b = blockIdx.z, h = blockIdx.y;
  int q0 = blockIdx.x * 64;
  size_t headoff = ((size_t)(b * H_ + h)) * S_ * DK_;
  const __bf16* Qp = q + headoff;
  const __bf16* Kp = k + headoff;
  const __bf16* Vp = v + headoff;
  int t = threadIdx.x, wave = t >> 5, lane = t & 31;
  int lm = lane & 15, g16 = lane >> 4;
  int hk = g16 * 8;

  const __bf16* qp = Qp + (size_t)(q0 + wave * 16 + lm) * DK_;
  v16bf qa0 = comb16(*(const v8bf*)(qp + 2 * hk),      *(const v8bf*)(qp + 8 + 2 * hk));
  v16bf qa1 = comb16(*(const v8bf*)(qp + 32 + 2 * hk), *(const v8bf*)(qp + 40 + 2 * hk));

  v8f o[4] = {};
  float mrun[8], lrun[8];
#pragma unroll
  for (int r = 0; r < 8; r++) { mrun[r] = -3.0e38f; lrun[r] = 0.0f; }

  for (int kt0 = 0; kt0 < S_; kt0 += 64) {
    __syncthreads();
    {  // stage V^T into LDS
      int key = t >> 1;
      int dc = (t & 1) * 32;
      const __bf16* vp = Vp + (size_t)(kt0 + key) * DK_ + dc;
#pragma unroll
      for (int c = 0; c < 4; ++c) {
        v8bf xv = *(const v8bf*)(vp + c * 8);
#pragma unroll
        for (int j = 0; j < 8; ++j) VTs[(dc + c * 8 + j) * 72 + key] = xv[j];
      }
    }
    __syncthreads();

    v8f s[4];
#pragma unroll
    for (int nt = 0; nt < 4; ++nt) {
      const __bf16* kp = Kp + (size_t)(kt0 + nt * 16 + lm) * DK_ + g16 * 16;
      v16bf b0 = comb16(*(const v8bf*)kp,        *(const v8bf*)(kp + 8));
      v16bf b1 = comb16(*(const v8bf*)(kp + 32), *(const v8bf*)(kp + 40));
      v8f cz = {};
      cz = wmma_bf16(qa0, b0, cz);
      cz = wmma_bf16(qa1, b1, cz);
      s[nt] = cz;
    }

#pragma unroll
    for (int r = 0; r < 8; ++r) {
      float mx = fmaxf(fmaxf(s[0][r], s[1][r]), fmaxf(s[2][r], s[3][r]));
      mx = redmax16(mx);
      float mnew = fmaxf(mrun[r], mx);
      float al = expf(mrun[r] - mnew);
      float ps = 0.0f;
#pragma unroll
      for (int nt = 0; nt < 4; ++nt) {
        float e = expf(s[nt][r] - mnew);
        s[nt][r] = e;
        ps += e;
      }
      ps = redsum16(ps);
      lrun[r] = lrun[r] * al + ps;
      mrun[r] = mnew;
#pragma unroll
      for (int d = 0; d < 4; ++d) o[d][r] = o[d][r] * al;
    }

    __bf16* pw = &Ps[wave][0];
#pragma unroll
    for (int nt = 0; nt < 4; ++nt)
#pragma unroll
      for (int r = 0; r < 8; ++r)
        pw[(g16 * 8 + r) * 72 + nt * 16 + lm] = (__bf16)s[nt][r];

#pragma unroll
    for (int ks = 0; ks < 2; ++ks) {
      const __bf16* pp = &Ps[wave][lm * 72 + ks * 32];
      v16bf pa = comb16(*(const v8bf*)(pp + 2 * hk), *(const v8bf*)(pp + 2 * hk + 8));
#pragma unroll
      for (int d = 0; d < 4; ++d) {
        const __bf16* vtp = &VTs[(d * 16 + lm) * 72 + ks * 32 + g16 * 16];
        v16bf vb = comb16(*(const v8bf*)vtp, *(const v8bf*)(vtp + 8));
        o[d] = wmma_bf16(pa, vb, o[d]);
      }
    }
  }

#pragma unroll
  for (int r = 0; r < 8; ++r) {
    float inv = 1.0f / lrun[r];
    int srow = q0 + wave * 16 + g16 * 8 + r;
    size_t tok = (size_t)b * S_ + srow;
#pragma unroll
    for (int d = 0; d < 4; ++d)
      ob[tok * (H_ * DK_) + h * DK_ + d * 16 + lm] = (__bf16)(o[d][r] * inv);
  }
}

// ---------------- workspace layout (bytes) ----------------
static const size_t OFF_HB    = 0;                       // bf16 [4096,1024]
static const size_t OFF_WQKVT = OFF_HB    + 8388608;     // bf16 [3072,1024]
static const size_t OFF_BQKV  = OFF_WQKVT + 6291456;     // f32  [3072]
static const size_t OFF_Q     = OFF_BQKV  + 12288;       // bf16 [B,H,S,DK]
static const size_t OFF_K     = OFF_Q     + 8388608;
static const size_t OFF_V     = OFF_K     + 8388608;
static const size_t OFF_OB    = OFF_V     + 8388608;     // bf16 [4096,1024]
static const size_t OFF_WOT   = OFF_OB    + 8388608;     // bf16 [1024,1024]
static const size_t OFF_X2    = OFF_WOT   + 2097152;     // f32  [4096,1024]
static const size_t OFF_H2B   = OFF_X2    + 16777216;    // bf16 [4096,1024]
static const size_t OFF_W1T   = OFF_H2B   + 8388608;     // bf16 [4096,1024]
static const size_t OFF_FFB   = OFF_W1T   + 8388608;     // bf16 [4096,4096]
static const size_t OFF_W2T   = OFF_FFB   + 33554432;    // bf16 [1024,4096]

extern "C" void kernel_launch(void* const* d_in, const int* in_sizes, int n_in,
                              void* d_out, int out_size, void* d_ws, size_t ws_size,
                              hipStream_t stream) {
  const float* x   = (const float*)d_in[0];
  const float* Wq  = (const float*)d_in[1];
  const float* bq  = (const float*)d_in[2];
  const float* Wk  = (const float*)d_in[3];
  const float* bk  = (const float*)d_in[4];
  const float* Wv  = (const float*)d_in[5];
  const float* bv  = (const float*)d_in[6];
  const float* Wo  = (const float*)d_in[7];
  const float* bo  = (const float*)d_in[8];
  const float* W1  = (const float*)d_in[9];
  const float* b1  = (const float*)d_in[10];
  const float* W2  = (const float*)d_in[11];
  const float* b2  = (const float*)d_in[12];
  const float* g1  = (const float*)d_in[13];
  const float* be1 = (const float*)d_in[14];
  const float* g2  = (const float*)d_in[15];
  const float* be2 = (const float*)d_in[16];
  float* out = (float*)d_out;

  char* ws = (char*)d_ws;
  __bf16* hb    = (__bf16*)(ws + OFF_HB);
  __bf16* wqkvT = (__bf16*)(ws + OFF_WQKVT);
  float*  bqkv  = (float*)(ws + OFF_BQKV);
  __bf16* qb    = (__bf16*)(ws + OFF_Q);
  __bf16* kb    = (__bf16*)(ws + OFF_K);
  __bf16* vb    = (__bf16*)(ws + OFF_V);
  __bf16* obuf  = (__bf16*)(ws + OFF_OB);
  __bf16* woT   = (__bf16*)(ws + OFF_WOT);
  float*  x2    = (float*)(ws + OFF_X2);
  __bf16* h2b   = (__bf16*)(ws + OFF_H2B);
  __bf16* w1T   = (__bf16*)(ws + OFF_W1T);
  __bf16* ffb   = (__bf16*)(ws + OFF_FFB);
  __bf16* w2T   = (__bf16*)(ws + OFF_W2T);

  conv_wqkv<<<(3072 * 1024) / 256, 256, 0, stream>>>(Wq, Wk, Wv, wqkvT);
  conv_bqkv<<<12, 256, 0, stream>>>(bq, bk, bv, bqkv);
  conv_T<<<(1024 * 1024) / 256, 256, 0, stream>>>(Wo, woT, 1024, 1024);
  conv_T<<<(4096 * 1024) / 256, 256, 0, stream>>>(W1, w1T, 4096, 1024);
  conv_T<<<(1024 * 4096) / 256, 256, 0, stream>>>(W2, w2T, 1024, 4096);

  ln_kernel<<<NTOK, 256, 0, stream>>>(x, g1, be1, hb);
  gemm_bf16<<<dim3(3072 / 128, NTOK / 64), 128, 0, stream>>>(
      hb, wqkvT, NTOK, 3072, 1024, 0, bqkv, nullptr, nullptr, nullptr, qb, kb, vb);
  fa_kernel<<<dim3(S_ / 64, H_, B_), 128, 0, stream>>>(qb, kb, vb, obuf);
  gemm_bf16<<<dim3(1024 / 128, NTOK / 64), 128, 0, stream>>>(
      obuf, woT, NTOK, 1024, 1024, 1, bo, x, x2, nullptr, nullptr, nullptr, nullptr);
  ln_kernel<<<NTOK, 256, 0, stream>>>(x2, g2, be2, h2b);
  gemm_bf16<<<dim3(4096 / 128, NTOK / 64), 128, 0, stream>>>(
      h2b, w1T, NTOK, 4096, 1024, 2, b1, nullptr, nullptr, ffb, nullptr, nullptr, nullptr);
  gemm_bf16<<<dim3(1024 / 128, NTOK / 64), 128, 0, stream>>>(
      ffb, w2T, NTOK, 1024, 4096, 3, b2, x2, out, nullptr, nullptr, nullptr, nullptr);
}
